// StochasticLSTMCell_13838384627838
// MI455X (gfx1250) — matrix operand order repaired and checked by reading, not compiled, not run on hardware
//
#include <hip/hip_runtime.h>

// ---------------------------------------------------------------------------
// Stochastic (concrete-dropout) LSTM for MI455X / gfx1250.
// T=512 sequential steps; batch-partitioned persistent kernel, bf16 WMMA.
// 8 workgroups (16 batch rows each) x 512 threads (16 wave32 waves).
// ---------------------------------------------------------------------------

typedef __attribute__((ext_vector_type(16))) __bf16 v16bf;
typedef __attribute__((ext_vector_type(8)))  float  v8f;

constexpr int Tn = 512;
constexpr int Bn = 128;
constexpr int Dn = 256;
constexpr int Hn = 512;
constexpr int BT = 16;           // batch rows per workgroup
constexpr int NTHREADS = 512;    // 16 waves (wave32)
constexpr float EPSC = 1e-7f;
constexpr float INV_TEMP = 10.0f;   // 1 / 0.1

// round-to-nearest-even f32 -> bf16 (stored as u16)
__device__ __forceinline__ unsigned short f2bf(float f) {
  unsigned int u = __float_as_uint(f);
  unsigned int r = u + 0x7FFFu + ((u >> 16) & 1u);
  return (unsigned short)(r >> 16);
}

__device__ __forceinline__ float fsig(float x) {
  return 1.0f / (1.0f + __expf(-x));
}
__device__ __forceinline__ float ftanh(float x) {
  float e = __expf(2.0f * x);
  return (e - 1.0f) / (e + 1.0f);
}

__device__ __forceinline__ v8f wmma_bf16(v16bf a, v16bf b, v8f c) {
  return __builtin_amdgcn_wmma_f32_16x16x32_bf16(
      false, a, false, b, (short)0, c, false, false);
}

// ---------------------------------------------------------------------------
// Prep 1: concrete-dropout masks zx (4,B,D) and zh (4,B,H), fp32.
// ---------------------------------------------------------------------------
__global__ void prep_masks(const float* __restrict__ ux,
                           const float* __restrict__ uh,
                           const float* __restrict__ p_logit,
                           float* __restrict__ zx, float* __restrict__ zh) {
  const float p  = 1.0f / (1.0f + __expf(-p_logit[0]));
  const float c0 = __logf(p + EPSC) - __logf(1.0f - p + EPSC);
  const float inv1mp = 1.0f / (1.0f - p);
  const int nx = 4 * Bn * Dn;
  const int nh = 4 * Bn * Hn;
  int i = blockIdx.x * blockDim.x + threadIdx.x;
  if (i < nx) {
    float u = ux[i];
    float l = (c0 + __logf(u + EPSC) - __logf(1.0f - u + EPSC)) * INV_TEMP;
    zx[i] = (1.0f - fsig(l)) * inv1mp;
  } else {
    int j = i - nx;
    if (j < nh) {
      float u = uh[j];
      float l = (c0 + __logf(u + EPSC) - __logf(1.0f - u + EPSC)) * INV_TEMP;
      zh[j] = (1.0f - fsig(l)) * inv1mp;
    }
  }
}

// ---------------------------------------------------------------------------
// Prep 2: swizzle Wx (4,H,D) and Wh (4,H,H) into bf16 WMMA-B fragments.
// Fragment = 32(K) x 16(N); lane = n + 16*(k/16); element e = k%16;
// stored at frag_base + lane*16 + e -> each lane loads 16 contiguous bf16.
// B[k][n] = W[g][n_out][k_in]  (gates = A(16xK) * W^T)
// ---------------------------------------------------------------------------
__global__ void prep_weights(const float* __restrict__ Wx,
                             const float* __restrict__ Wh,
                             unsigned short* __restrict__ Wxb,
                             unsigned short* __restrict__ Whb) {
  const int nwx = 4 * Hn * Dn;
  const int nwh = 4 * Hn * Hn;
  int i = blockIdx.x * blockDim.x + threadIdx.x;
  if (i < nwx) {
    int within = i & 511, fidx = i >> 9;
    int lane = within >> 4, e = within & 15;
    int n = lane & 15, k = ((lane >> 4) << 4) + e;
    int kt = fidx & 7, nt = (fidx >> 3) & 31, g = fidx >> 8;
    Wxb[i] = f2bf(Wx[((size_t)g * Hn + nt * 16 + n) * Dn + kt * 32 + k]);
  } else {
    int j = i - nwx;
    if (j < nwh) {
      int within = j & 511, fidx = j >> 9;
      int lane = within >> 4, e = within & 15;
      int n = lane & 15, k = ((lane >> 4) << 4) + e;
      int kt = fidx & 15, nt = (fidx >> 4) & 31, g = fidx >> 9;
      Whb[j] = f2bf(Wh[((size_t)g * Hn + nt * 16 + n) * Hn + kt * 32 + k]);
    }
  }
}

// ---------------------------------------------------------------------------
// Main persistent LSTM scan.
// ---------------------------------------------------------------------------
__global__ __launch_bounds__(NTHREADS, 1)
void lstm_scan(const float* __restrict__ x, const float* __restrict__ zx,
               const float* __restrict__ zh,
               const unsigned short* __restrict__ Wxb,
               const unsigned short* __restrict__ Whb,
               const float* __restrict__ bx, const float* __restrict__ bh,
               float* __restrict__ out) {
  __shared__ __align__(64) float          hs[BT * Hn];            // 32 KB
  __shared__ __align__(64) unsigned short AxL[4 * 8 * 512];       // 32 KB
  __shared__ __align__(64) unsigned short AhL[4 * 16 * 512];      // 64 KB

  const int tid  = threadIdx.x;
  const int lane = tid & 31;
  const int wave = tid >> 5;
  const int b0   = blockIdx.x * BT;
  const int n    = lane & 15;            // C-tile column
  const int mhi  = (lane >> 4) << 3;     // C rows: lanes 16..31 -> M+8

  // Per-thread A-build swizzle constants (tid<512 => slot 'within' == tid).
  const int e_a = tid & 15;
  const int ln  = tid >> 4;              // fragment lane 0..31
  const int m_a = ln & 15;               // A row this thread writes
  const int kk  = ((ln >> 4) << 4) + e_a;

  for (int i = tid; i < BT * Hn; i += NTHREADS) hs[i] = 0.0f;
  float creg[2][8];
#pragma unroll
  for (int a = 0; a < 2; ++a)
#pragma unroll
    for (int j = 0; j < 8; ++j) creg[a][j] = 0.0f;

  // Bias sums are t-invariant: hoist (2 N-tiles x 4 gates).
  const int nt0 = wave << 1;
  float bsum[2][4];
#pragma unroll
  for (int p = 0; p < 2; ++p) {
    int kg = ((nt0 + p) << 4) + n;
#pragma unroll
    for (int g = 0; g < 4; ++g) bsum[p][g] = bx[g * Hn + kg] + bh[g * Hn + kg];
  }
  __syncthreads();

  for (int t = 0; t < Tn; ++t) {
    // Opaque zero offset: keeps loads loop-variant (no LICM hoisting/spill)
    // while the GEP base stays the kernarg pointer, so address-space
    // inference still emits global_load_b128 (LOADcnt-only, no FLAT/DScnt).
    size_t opq = 0;
    asm volatile("" : "+s"(opq));
    const float* zx_t = zx + opq;
    const float* zh_t = zh + opq;
    const unsigned short* Wxb_t = Wxb + opq;
    const unsigned short* Whb_t = Whb + opq;

    // Prefetch next timestep's input rows (16 x 256 f32 = 16 KB) into WGP$.
    if (t + 1 < Tn && tid < 256)
      __builtin_prefetch((const char*)&x[((size_t)(t + 1) * Bn + b0) * Dn] + tid * 64,
                         0, 3);

    // ---- build masked bf16 A operands in LDS (contiguous k = 16h + s, same order as B) ----
    {
      const size_t xrow = ((size_t)t * Bn + b0 + m_a) * Dn;
#pragma unroll
      for (int s = 0; s < 32; ++s) {               // Ax: g = s/8, kt = s%8
        int g = s >> 3, col = ((s & 7) << 5) + kk;
        float v = x[xrow + col] * zx_t[((g * Bn) + b0 + m_a) * Dn + col];
        AxL[(s << 9) + tid] = f2bf(v);
      }
#pragma unroll
      for (int s = 0; s < 64; ++s) {               // Ah: g = s/16, kt = s%16
        int g = s >> 4, col = ((s & 15) << 5) + kk;
        float v = hs[m_a * Hn + col] * zh_t[((g * Bn) + b0 + m_a) * Hn + col];
        AhL[(s << 9) + tid] = f2bf(v);
      }
    }
    __syncthreads();

    // ---- 96-tile GEMM chain: A loaded once, 2 WMMAs (two N-tiles) each ----
    // u = g*24 + r ; r<8 -> input-proj K-tile, r>=8 -> hidden-proj K-tile.
    v8f acc[8];                                    // [gate][nti]
#pragma unroll
    for (int q = 0; q < 8; ++q) acc[q] = (v8f){0, 0, 0, 0, 0, 0, 0, 0};

    auto loadA = [&](int g, int r) -> v16bf {
      if (r < 8) return *(const v16bf*)&AxL[(((g << 3) + r) << 9) + (lane << 4)];
      return *(const v16bf*)&AhL[(((g << 4) + (r - 8)) << 9) + (lane << 4)];
    };
    auto loadB = [&](int g, int r, int nt) -> v16bf {
      if (r < 8)
        return *(const v16bf*)
            &Wxb_t[((size_t)(((g << 5) + nt) << 3) + r) * 512 + (lane << 4)];
      return *(const v16bf*)
          &Whb_t[((size_t)(((g << 5) + nt) << 4) + (r - 8)) * 512 + (lane << 4)];
    };

    v16bf a_c  = loadA(0, 0);
    v16bf b0_c = loadB(0, 0, nt0);
    v16bf b1_c = loadB(0, 0, nt0 + 1);
#pragma unroll
    for (int u = 0; u < 96; ++u) {
      const int g = u / 24;
      v16bf a_n, b0_n, b1_n;
      if (u + 1 < 96) {                            // double-buffer next tile
        const int g2 = (u + 1) / 24, r2 = (u + 1) % 24;
        a_n  = loadA(g2, r2);
        b0_n = loadB(g2, r2, nt0);
        b1_n = loadB(g2, r2, nt0 + 1);
      }
      acc[g * 2 + 0] = wmma_bf16(a_c, b0_c, acc[g * 2 + 0]);
      acc[g * 2 + 1] = wmma_bf16(a_c, b1_c, acc[g * 2 + 1]);
      a_c = a_n; b0_c = b0_n; b1_c = b1_n;
    }

    // ---- LSTM cell update (all 4 gates wave-local, element-aligned) ----
#pragma unroll
    for (int p = 0; p < 2; ++p) {
      const int kg = ((nt0 + p) << 4) + n;
#pragma unroll
      for (int j = 0; j < 8; ++j) {
        int m = mhi + j;
        float gi = fsig(acc[0 * 2 + p][j] + bsum[p][0]);
        float gf = fsig(acc[1 * 2 + p][j] + bsum[p][1]);
        float go = fsig(acc[2 * 2 + p][j] + bsum[p][2]);
        float gg = ftanh(acc[3 * 2 + p][j] + bsum[p][3]);
        float c  = gf * creg[p][j] + gi * gg;
        creg[p][j] = c;
        float hv = go * ftanh(c);
        hs[m * Hn + kg] = hv;
        out[((size_t)t * Bn + (b0 + m)) * Hn + kg] = hv;   // hn
      }
    }
    __syncthreads();
  }

  // ---- final h_T, c_T ----
  const size_t base_h = (size_t)Tn * Bn * Hn;
  const size_t base_c = base_h + (size_t)Bn * Hn;
  for (int i = tid; i < BT * Hn; i += NTHREADS) {
    int m = i >> 9, k = i & (Hn - 1);
    out[base_h + (size_t)(b0 + m) * Hn + k] = hs[i];
  }
#pragma unroll
  for (int p = 0; p < 2; ++p) {
    const int kg = ((nt0 + p) << 4) + n;
#pragma unroll
    for (int j = 0; j < 8; ++j) {
      int m = mhi + j;
      out[base_c + (size_t)(b0 + m) * Hn + kg] = creg[p][j];
    }
  }
}

// ---------------------------------------------------------------------------
extern "C" void kernel_launch(void* const* d_in, const int* in_sizes, int n_in,
                              void* d_out, int out_size, void* d_ws, size_t ws_size,
                              hipStream_t stream) {
  const float* input   = (const float*)d_in[0];   // (T,B,D)
  const float* ux      = (const float*)d_in[1];   // (4,B,D)
  const float* uh      = (const float*)d_in[2];   // (4,B,H)
  const float* p_logit = (const float*)d_in[3];   // (1,)
  const float* Wx      = (const float*)d_in[4];   // (4,H,D)
  const float* bx      = (const float*)d_in[5];   // (4,H)
  const float* Wh      = (const float*)d_in[6];   // (4,H,H)
  const float* bh      = (const float*)d_in[7];   // (4,H)
  float* out = (float*)d_out;

  // workspace (~4.5 MB): zx | zh | Wxb(bf16) | Whb(bf16)
  float* zx = (float*)d_ws;
  float* zh = zx + 4 * Bn * Dn;
  unsigned short* Wxb = (unsigned short*)(zh + 4 * Bn * Hn);
  unsigned short* Whb = Wxb + 4 * Hn * Dn;

  const int nmask = 4 * Bn * Dn + 4 * Bn * Hn;
  prep_masks<<<(nmask + 255) / 256, 256, 0, stream>>>(ux, uh, p_logit, zx, zh);

  const int nw = 4 * Hn * Dn + 4 * Hn * Hn;
  prep_weights<<<(nw + 255) / 256, 256, 0, stream>>>(Wx, Wh, Wxb, Whb);

  lstm_scan<<<Bn / BT, NTHREADS, 0, stream>>>(input, zx, zh, Wxb, Whb, bx, bh, out);
}
